// SelfAttention_D_13451837571123
// MI455X (gfx1250) — matrix operand order repaired and checked by reading, not compiled, not run on hardware
//
#include <hip/hip_runtime.h>
#include <hip/hip_bf16.h>

// ---------------------------------------------------------------------------
// MI455X (gfx1250) full-model forward:
//   conv1d stack (fused BN+LeakyReLU) -> SAGAN self-attn -> c3c -> mean ->
//   proj -> GAT(body/hand) -> out-proj -> tiled concat -> k3 logits conv.
// All GEMM-shaped work runs through v_wmma_f32_16x16x32_f16 (wave32 WMMA):
// 128x128 block tile, 8 waves x (2x4 16x16 tiles), K step 32, double-buffered
// LDS staging (one workgroup barrier per K step), branch-free clamped loads.
// Layouts follow CDNA5 ISA 7.12.2.
// ---------------------------------------------------------------------------

typedef _Float16 h16;
typedef __attribute__((ext_vector_type(16))) _Float16 v16h;
typedef __attribute__((ext_vector_type(8)))  float    v8f;

struct GemmDesc {
  int M, N, K, nbatch;
  long aBS; int aSM, aSK;          // A element (m,k): A[z*aBS + m*aSM + k*aSK]
  long bBS; int bSK, bSN;          // B element (k,n) (plain GEMM mode, KW==0)
  int conv_stride, conv_pad, Tin;  // im2col mode (KW>0): B = (nbatch, Cin, Tin)
  long oBS; int oSM;               // out element (m,n): O[z*oBS + m*oSM + n]
  int epi;                         // 0 none, 1 +bias[m], 2 +bias[m]+BN+lrelu,
                                   // 3 gamma*acc+resid, 4 +bias[n]
};

__constant__ int c_body_par[10] = {-1,0,1,2,3,1,5,6,1,8};
__constant__ int c_hand_par[42] = {-1,0,1,2,3,0,5,6,7,0,9,10,11,0,13,14,15,0,17,18,19,
                                   -1,21,22,23,24,21,26,27,28,21,30,31,32,21,34,35,36,
                                   21,38,39,40};

// ---------------------------------------------------------------------------
// Generic WMMA GEMM. KW: 0 = plain strided GEMM B; 3/4 = fused im2col conv1d.
// ---------------------------------------------------------------------------
template <typename TA, typename TB, typename TO, int KW>
__global__ __launch_bounds__(256)
void wmma_gemm_kernel(GemmDesc d, const TA* __restrict__ A, const TB* __restrict__ B,
                      TO* __restrict__ O, const float* __restrict__ bias,
                      const float* __restrict__ g, const float* __restrict__ be,
                      const float* __restrict__ mn, const float* __restrict__ vr,
                      const float* __restrict__ gammaPtr, const h16* __restrict__ resid)
{
  __shared__ h16 As[2][128][40];   // [buf][m][k]
  __shared__ h16 Bt[2][128][40];   // [buf][n][k]  (transposed for b128 frag reads)
  const int tid  = threadIdx.x;
  const int lane = tid & 31;
  const int wave = tid >> 5;
  const int wm = (wave >> 1) << 5;     // 0,32,64,96
  const int wn = (wave & 1) << 6;      // 0,64
  const int z  = blockIdx.z;
  const int m0 = blockIdx.y << 7;
  const int n0 = blockIdx.x << 7;

  v8f acc[2][4] = {};

  // branch-free staging: clamp OOB addresses to 0 and select the value
  auto stageA = [&](int k0, int buf) {
#pragma unroll
    for (int i = 0; i < 16; ++i) {
      int e = (i << 8) + tid;
      int r = e >> 5, c = e & 31;
      int gm = m0 + r, gk = k0 + c;
      bool ok = (gm < d.M) && (gk < d.K);
      long idx = (long)z * d.aBS + (long)gm * d.aSM + (long)gk * d.aSK;
      float v = (float)A[ok ? idx : 0];
      As[buf][r][c] = (h16)(ok ? v : 0.f);
    }
  };
  auto stageB = [&](int k0, int buf) {
#pragma unroll
    for (int i = 0; i < 16; ++i) {
      int e = (i << 8) + tid;
      int r = e >> 7, c = e & 127;     // r = k (0..31), c = n (0..127)
      int gk = k0 + r, gn = n0 + c;
      bool ok = (gk < d.K) && (gn < d.N);
      long idx;
      if (KW > 0) {                    // im2col: compile-time kernel width
        int ci  = gk / KW;
        int kwi = gk - ci * KW;
        int tin = gn * d.conv_stride - d.conv_pad + kwi;
        ok = ok && (tin >= 0) && (tin < d.Tin);
        idx = (long)z * d.bBS + (long)ci * d.Tin + tin;
      } else {
        idx = (long)z * d.bBS + (long)gk * d.bSK + (long)gn * d.bSN;
      }
      float v = (float)B[ok ? idx : 0];
      Bt[buf][c][r] = (h16)(ok ? v : 0.f);
    }
  };

  const int ksteps = (d.K + 31) >> 5;
  stageA(0, 0);
  stageB(0, 0);
  int buf = 0;
  for (int s = 0; s < ksteps; ++s) {
    __syncthreads();                   // staging of `buf` complete; prev compute done
    if (s + 1 < ksteps) {              // overlap next tile's global loads with WMMA
      stageA((s + 1) << 5, buf ^ 1);
      stageB((s + 1) << 5, buf ^ 1);
    }
    // fragments (ISA 7.12.2 16-bit A 16x32 layout; B via transposed LDS tile)
    v16h af[2], bf[4];
    const int rsel = lane & 15;
    const int akb  = (lane < 16) ? 0 : 8;
    const int bkb  = (lane < 16) ? 0 : 8;
#pragma unroll
    for (int ti = 0; ti < 2; ++ti) {
      const h16* row = &As[buf][wm + ti * 16 + rsel][0];
#pragma unroll
      for (int j = 0; j < 8; ++j) af[ti][j]     = row[akb + j];
#pragma unroll
      for (int j = 0; j < 8; ++j) af[ti][8 + j] = row[akb + 16 + j];
    }
#pragma unroll
    for (int tj = 0; tj < 4; ++tj) {
      const h16* colp = &Bt[buf][wn + tj * 16 + rsel][0];
#pragma unroll
      for (int j = 0; j < 8; ++j) bf[tj][j]     = colp[bkb + j];
#pragma unroll
      for (int j = 0; j < 8; ++j) bf[tj][8 + j] = colp[bkb + 16 + j];
    }
#pragma unroll
    for (int ti = 0; ti < 2; ++ti)
#pragma unroll
      for (int tj = 0; tj < 4; ++tj)
        acc[ti][tj] = __builtin_amdgcn_wmma_f32_16x16x32_f16(
            false, af[ti], false, bf[tj], (short)0, acc[ti][tj], false, false);
    buf ^= 1;
  }

  // epilogue: C/D layout => m = vgpr + (lane>=16 ? 8 : 0), n = lane&15
  const float gm2 = (d.epi == 3) ? gammaPtr[0] : 0.f;
#pragma unroll
  for (int ti = 0; ti < 2; ++ti) {
#pragma unroll
    for (int tj = 0; tj < 4; ++tj) {
      int mb = m0 + wm + ti * 16 + ((lane >> 4) << 3);
      int nn = n0 + wn + tj * 16 + (lane & 15);
      if (nn >= d.N) continue;
#pragma unroll
      for (int vv = 0; vv < 8; ++vv) {
        int m = mb + vv;
        if (m >= d.M) continue;
        float r = acc[ti][tj][vv];
        if (d.epi == 1) {
          r += bias[m];
        } else if (d.epi == 2) {
          r += bias[m];
          r = g[m] * (r - mn[m]) * rsqrtf(vr[m] + 1e-5f) + be[m];
          r = (r >= 0.f) ? r : 0.2f * r;
        } else if (d.epi == 3) {
          r = gm2 * r + (float)resid[(long)z * d.oBS + (long)m * d.oSM + nn];
        } else if (d.epi == 4) {
          r += bias[nn];
        }
        O[(long)z * d.oBS + (long)m * d.oSM + nn] = (TO)r;
      }
    }
  }
}

// ---------------------------------------------------------------------------
// Small helper kernels
// ---------------------------------------------------------------------------
__global__ void xpose_kernel(const float* __restrict__ X, h16* __restrict__ O, int n) {
  int i = blockIdx.x * 256 + threadIdx.x;
  if (i >= n) return;
  int t = i & 2047;
  int rest = i >> 11;
  int c = rest % 104;
  int b = rest / 104;
  O[i] = (h16)X[((long)b * 2048 + t) * 104 + c];
}

__global__ __launch_bounds__(256)
void softmax_kernel(const float* __restrict__ E, h16* __restrict__ A, int T) {
  long row = blockIdx.x;
  const float* e = E + row * T;
  h16* a = A + row * T;
  __shared__ float sd[256];
  int tid = threadIdx.x;
  float mx = -3.0e38f;
  for (int j = tid; j < T; j += 256) mx = fmaxf(mx, e[j]);
  sd[tid] = mx; __syncthreads();
  for (int s = 128; s > 0; s >>= 1) { if (tid < s) sd[tid] = fmaxf(sd[tid], sd[tid + s]); __syncthreads(); }
  mx = sd[0]; __syncthreads();
  float sum = 0.f;
  for (int j = tid; j < T; j += 256) sum += __expf(e[j] - mx);
  sd[tid] = sum; __syncthreads();
  for (int s = 128; s > 0; s >>= 1) { if (tid < s) sd[tid] += sd[tid + s]; __syncthreads(); }
  float den = sd[0] + 1e-16f;
  for (int j = tid; j < T; j += 256) a[j] = (h16)(__expf(e[j] - mx) / den);
}

__global__ void mean_t_kernel(const h16* __restrict__ X, h16* __restrict__ O, int rows, int T) {
  int i = blockIdx.x * 256 + threadIdx.x;
  if (i >= rows) return;
  const h16* p = X + (long)i * T;
  float s = 0.f;
  for (int t = 0; t < T; ++t) s += (float)p[t];
  O[i] = (h16)(s / (float)T);
}

__global__ void att_score_kernel(const h16* __restrict__ X, const float* __restrict__ asrc,
                                 const float* __restrict__ adst, float* __restrict__ AS,
                                 float* __restrict__ AD, int rows) {
  int i = blockIdx.x * 256 + threadIdx.x;
  if (i >= rows * 4) return;
  int row = i >> 2, h = i & 3;
  float s = 0.f, dd = 0.f;
  for (int f = 0; f < 64; ++f) {
    float xv = (float)X[(long)row * 256 + h * 64 + f];
    s  += xv * asrc[h * 64 + f];
    dd += xv * adst[h * 64 + f];
  }
  AS[i] = s; AD[i] = dd;
}

__global__ __launch_bounds__(64)
void gat_agg_kernel(const h16* __restrict__ X, const float* __restrict__ AS,
                    const float* __restrict__ AD, const float* __restrict__ bias,
                    h16* __restrict__ O, int Nn, int which) {
  int b = blockIdx.x, i = blockIdx.y, f = threadIdx.x;
  int pi = which ? c_hand_par[i] : c_body_par[i];
  float acc = 0.f;
  for (int h = 0; h < 4; ++h) {
    float adi = AD[((long)b * Nn + i) * 4 + h];
    float mx = -3.0e38f;
    for (int j = 0; j < Nn; ++j) {
      int pj = which ? c_hand_par[j] : c_body_par[j];
      if (i == j || pi == j || pj == i) {
        float e = AS[((long)b * Nn + j) * 4 + h] + adi;
        e = (e >= 0.f) ? e : 0.2f * e;
        mx = fmaxf(mx, e);
      }
    }
    float den = 0.f, out = 0.f;
    for (int j = 0; j < Nn; ++j) {
      int pj = which ? c_hand_par[j] : c_body_par[j];
      if (i == j || pi == j || pj == i) {
        float e = AS[((long)b * Nn + j) * 4 + h] + adi;
        e = (e >= 0.f) ? e : 0.2f * e;
        den += __expf(e - mx);
      }
    }
    for (int j = 0; j < Nn; ++j) {
      int pj = which ? c_hand_par[j] : c_body_par[j];
      if (i == j || pi == j || pj == i) {
        float e = AS[((long)b * Nn + j) * 4 + h] + adi;
        e = (e >= 0.f) ? e : 0.2f * e;
        float al = __expf(e - mx) / (den + 1e-16f);
        out += al * (float)X[((long)b * Nn + j) * 256 + h * 64 + f];
      }
    }
    acc += out;
  }
  O[((long)b * Nn + i) * 64 + f] = (h16)(acc * 0.25f + bias[f]);
}

__global__ __launch_bounds__(256)
void logits_kernel(const h16* __restrict__ A9, const h16* __restrict__ XG,
                   const float* __restrict__ W, const float* __restrict__ bias,
                   float* __restrict__ out, int T) {
  int bt = blockIdx.x;
  int b = bt / T, t = bt - b * T;
  int tid = threadIdx.x;
  bool v0 = (t - 1) >= 0, v2 = (t + 1) < T;
  float s = 0.f;
  for (int c = tid; c < 2048; c += 256) {
    const float* w0 = W + (long)c * 3;
    const h16* ap = A9 + ((long)b * 2048 + c) * T + t;
    if (v0) s += (float)ap[-1] * w0[0];
    s += (float)ap[0] * w0[1];
    if (v2) s += (float)ap[1] * w0[2];
    const float* w1 = W + (long)(2048 + c) * 3;
    float wsum = w1[1] + (v0 ? w1[0] : 0.f) + (v2 ? w1[2] : 0.f);
    s += (float)XG[(long)b * 2048 + c] * wsum;
  }
  __shared__ float sd[256];
  sd[tid] = s; __syncthreads();
  for (int r = 128; r > 0; r >>= 1) { if (tid < r) sd[tid] += sd[tid + r]; __syncthreads(); }
  if (tid == 0) out[bt] = sd[0] + bias[0];
}

// ---------------------------------------------------------------------------
template <typename TA, typename TB, typename TO, int KW>
static void launch_gemm(hipStream_t s, const GemmDesc& d, const void* A, const void* B,
                        void* O, const float* bias = nullptr, const float* g = nullptr,
                        const float* be = nullptr, const float* mn = nullptr,
                        const float* vr = nullptr, const float* gamma = nullptr,
                        const void* resid = nullptr) {
  dim3 grid((unsigned)((d.N + 127) / 128), (unsigned)((d.M + 127) / 128), (unsigned)d.nbatch);
  wmma_gemm_kernel<TA, TB, TO, KW><<<grid, 256, 0, s>>>(
      d, (const TA*)A, (const TB*)B, (TO*)O, bias, g, be, mn, vr, gamma, (const h16*)resid);
}

struct BnP { const float *b, *be, *g, *m, *v, *w; };

extern "C" void kernel_launch(void* const* d_in, const int* in_sizes, int n_in,
                              void* d_out, int out_size, void* d_ws, size_t ws_size,
                              hipStream_t stream) {
  (void)in_sizes; (void)out_size;
  if (n_in < 80) return;

  // Input flattening assumed jax-pytree order (sorted dict keys, recursive).
  const float* bgat_W   = (const float*)d_in[0];
  const float* bgat_ad  = (const float*)d_in[1];
  const float* bgat_as  = (const float*)d_in[2];
  const float* bgat_b   = (const float*)d_in[3];
  const float* body_out_b  = (const float*)d_in[4];
  const float* body_out_w  = (const float*)d_in[5];
  const float* body_proj_b = (const float*)d_in[6];
  const float* body_proj_w = (const float*)d_in[7];
  const float* hgat_W   = (const float*)d_in[62];
  const float* hgat_ad  = (const float*)d_in[63];
  const float* hgat_as  = (const float*)d_in[64];
  const float* hgat_b   = (const float*)d_in[65];
  const float* hand_out_b  = (const float*)d_in[66];
  const float* hand_out_w  = (const float*)d_in[67];
  const float* hand_proj_b = (const float*)d_in[68];
  const float* hand_proj_w = (const float*)d_in[69];
  const float* logits_b = (const float*)d_in[70];
  const float* logits_w = (const float*)d_in[71];
  const float* sa_bk = (const float*)d_in[72];
  const float* sa_bq = (const float*)d_in[73];
  const float* sa_bv = (const float*)d_in[74];
  const float* sa_gamma = (const float*)d_in[75];
  const float* sa_wk = (const float*)d_in[76];
  const float* sa_wq = (const float*)d_in[77];
  const float* sa_wv = (const float*)d_in[78];
  const float* xin = (const float*)d_in[79];

  auto bn = [&](int base) {
    return BnP{(const float*)d_in[base + 0], (const float*)d_in[base + 1],
               (const float*)d_in[base + 2], (const float*)d_in[base + 3],
               (const float*)d_in[base + 4], (const float*)d_in[base + 5]};
  };

  const int Bz = 32, T8 = 252;
  // workspace carve-out (f16 unless noted)
  char* ws = (char*)d_ws;
  size_t off = 0;
  auto alloc = [&](size_t bytes) { size_t r = off; off += (bytes + 255) & ~(size_t)255; return r; };
  const size_t BIGH = (size_t)Bz * 2048 * T8;               // 16.5M halfs
  size_t oXT  = alloc((size_t)Bz * 104 * 2048 * 2);
  size_t oB0  = alloc(BIGH * 2);
  size_t oB1  = alloc(BIGH * 2);
  size_t oB2  = alloc(BIGH * 2);
  size_t oQ   = alloc((size_t)Bz * 256 * T8 * 2);
  size_t oK   = alloc((size_t)Bz * 256 * T8 * 2);
  size_t oE   = alloc((size_t)Bz * T8 * T8 * 4);            // f32
  size_t oATT = alloc((size_t)Bz * T8 * T8 * 2);
  size_t oMEAN= alloc((size_t)Bz * 2048 * 2);
  size_t oPB  = alloc((size_t)Bz * 640 * 2);
  size_t oPH  = alloc((size_t)Bz * 2688 * 2);
  size_t oGXB = alloc((size_t)Bz * 10 * 256 * 2);
  size_t oGXH = alloc((size_t)Bz * 42 * 256 * 2);
  size_t oASB = alloc((size_t)Bz * 10 * 4 * 4);
  size_t oADB = alloc((size_t)Bz * 10 * 4 * 4);
  size_t oASH = alloc((size_t)Bz * 42 * 4 * 4);
  size_t oADH = alloc((size_t)Bz * 42 * 4 * 4);
  size_t oGOB = alloc((size_t)Bz * 640 * 2);
  size_t oGOH = alloc((size_t)Bz * 2688 * 2);
  size_t oXG  = alloc((size_t)Bz * 2048 * 2);
  if (off > ws_size) return;  // scratch too small; nothing safe to do

  h16*  XT  = (h16*)(ws + oXT);
  h16*  BIG0= (h16*)(ws + oB0);
  h16*  BIG1= (h16*)(ws + oB1);
  h16*  BIG2= (h16*)(ws + oB2);
  h16*  Qb  = (h16*)(ws + oQ);
  h16*  Kb  = (h16*)(ws + oK);
  float* Eb = (float*)(ws + oE);
  h16*  ATT = (h16*)(ws + oATT);
  h16*  MEAN= (h16*)(ws + oMEAN);
  h16*  PB  = (h16*)(ws + oPB);
  h16*  PH  = (h16*)(ws + oPH);
  h16*  GXB = (h16*)(ws + oGXB);
  h16*  GXH = (h16*)(ws + oGXH);
  float* ASB= (float*)(ws + oASB);
  float* ADB= (float*)(ws + oADB);
  float* ASH= (float*)(ws + oASH);
  float* ADH= (float*)(ws + oADH);
  h16*  GOB = (h16*)(ws + oGOB);
  h16*  GOH = (h16*)(ws + oGOH);
  h16*  XG  = (h16*)(ws + oXG);

  auto conv = [&](const h16* in, h16* out, BnP p, int Cin, int Cout, int Tin, int Tout,
                  int kw, int stride, int pad) {
    GemmDesc d{};
    d.M = Cout; d.N = Tout; d.K = Cin * kw; d.nbatch = Bz;
    d.aBS = 0; d.aSM = d.K; d.aSK = 1;
    d.bBS = (long)Cin * Tin;
    d.conv_stride = stride; d.conv_pad = pad; d.Tin = Tin;
    d.oBS = (long)Cout * Tout; d.oSM = Tout; d.epi = 2;
    if (kw == 4)
      launch_gemm<float, h16, h16, 4>(stream, d, p.w, in, out, p.b, p.g, p.be, p.m, p.v);
    else
      launch_gemm<float, h16, h16, 3>(stream, d, p.w, in, out, p.b, p.g, p.be, p.m, p.v);
  };
  auto conv1x1 = [&](const float* W, const float* biasM, const h16* in, h16* out, int Cout) {
    GemmDesc d{};
    d.M = Cout; d.N = T8; d.K = 2048; d.nbatch = Bz;
    d.aBS = 0; d.aSM = 2048; d.aSK = 1;
    d.bBS = (long)2048 * T8; d.bSK = T8; d.bSN = 1;
    d.oBS = (long)Cout * T8; d.oSM = T8; d.epi = 1;
    launch_gemm<float, h16, h16, 0>(stream, d, W, in, out, biasM);
  };
  auto dense = [&](const h16* Ap, int aSM, const float* W, int ldw, const float* biasN,
                   h16* Op, int oSM, int M, int N, int K) {
    GemmDesc d{};
    d.M = M; d.N = N; d.K = K; d.nbatch = 1;
    d.aBS = 0; d.aSM = aSM; d.aSK = 1;
    d.bBS = 0; d.bSK = ldw; d.bSN = 1;
    d.oBS = 0; d.oSM = oSM; d.epi = biasN ? 4 : 0;
    launch_gemm<h16, float, h16, 0>(stream, d, Ap, W, Op, biasN);
  };

  // 1) (B,T,104) f32 -> (B,104,2048) f16
  {
    int n = Bz * 104 * 2048;
    xpose_kernel<<<(n + 255) / 256, 256, 0, stream>>>(xin, XT, n);
  }
  // 2) conv stack (rotating BIG buffers)
  conv(XT,   BIG1, bn(8),  104,   64, 2048, 1024, 4, 2, 1);  // c1a
  conv(BIG1, BIG2, bn(14),  64,   64, 1024, 1023, 4, 1, 1);  // c1b
  conv(BIG2, BIG0, bn(20),  64,  128, 1023,  511, 4, 2, 1);  // c2a1
  conv(BIG0, BIG1, bn(26), 128,  128,  511,  510, 4, 1, 1);  // c2a2
  conv(BIG1, BIG2, bn(32), 128,  512,  510,  255, 4, 2, 1);  // c2b1
  conv(BIG2, BIG0, bn(38), 512,  512,  255,  254, 4, 1, 1);  // c2b2
  conv(BIG0, BIG1, bn(44), 512, 1024,  254,  253, 4, 1, 1);  // c3a
  conv(BIG1, BIG2, bn(50),1024, 2048,  253,  252, 4, 1, 1);  // c3b -> a8 = BIG2

  // 3) self-attention
  conv1x1(sa_wq, sa_bq, BIG2, Qb, 256);
  conv1x1(sa_wk, sa_bk, BIG2, Kb, 256);
  conv1x1(sa_wv, sa_bv, BIG2, BIG0, 2048);                    // v -> BIG0
  {
    GemmDesc d{};                                             // energy = q^T k
    d.M = T8; d.N = T8; d.K = 256; d.nbatch = Bz;
    d.aBS = (long)256 * T8; d.aSM = 1;  d.aSK = T8;
    d.bBS = (long)256 * T8; d.bSK = T8; d.bSN = 1;
    d.oBS = (long)T8 * T8; d.oSM = T8; d.epi = 0;
    launch_gemm<h16, h16, float, 0>(stream, d, Qb, Kb, Eb);
  }
  softmax_kernel<<<Bz * T8, 256, 0, stream>>>(Eb, ATT, T8);
  {
    GemmDesc d{};                                             // sa_out = g*(V attn^T)+a8
    d.M = 2048; d.N = T8; d.K = T8; d.nbatch = Bz;
    d.aBS = (long)2048 * T8; d.aSM = T8; d.aSK = 1;
    d.bBS = (long)T8 * T8;   d.bSK = 1;  d.bSN = T8;
    d.oBS = (long)2048 * T8; d.oSM = T8; d.epi = 3;
    launch_gemm<h16, h16, h16, 0>(stream, d, BIG0, ATT, BIG1, nullptr, nullptr, nullptr,
                                  nullptr, nullptr, sa_gamma, BIG2);
  }
  conv(BIG1, BIG0, bn(56), 2048, 2048, 252, 252, 3, 1, 1);    // c3c -> a9 = BIG0

  // 4) channel means -> proj -> GAT -> out-proj -> XG
  mean_t_kernel<<<(Bz * 2048 + 255) / 256, 256, 0, stream>>>(BIG0, MEAN, Bz * 2048, T8);
  dense(MEAN,        2048, body_proj_w,  640, body_proj_b, PB,  640, Bz,  640, 1024);
  dense(MEAN + 1024, 2048, hand_proj_w, 2688, hand_proj_b, PH, 2688, Bz, 2688, 1024);
  dense(PB, 64, bgat_W, 256, nullptr, GXB, 256, Bz * 10, 256, 64);
  dense(PH, 64, hgat_W, 256, nullptr, GXH, 256, Bz * 42, 256, 64);
  att_score_kernel<<<(Bz * 10 * 4 + 255) / 256, 256, 0, stream>>>(GXB, bgat_as, bgat_ad, ASB, ADB, Bz * 10);
  att_score_kernel<<<(Bz * 42 * 4 + 255) / 256, 256, 0, stream>>>(GXH, hgat_as, hgat_ad, ASH, ADH, Bz * 42);
  gat_agg_kernel<<<dim3(Bz, 10), 64, 0, stream>>>(GXB, ASB, ADB, bgat_b, GOB, 10, 0);
  gat_agg_kernel<<<dim3(Bz, 42), 64, 0, stream>>>(GXH, ASH, ADH, hgat_b, GOH, 42, 1);
  dense(GOB,  640, body_out_w, 1024, body_out_b, XG,        2048, Bz, 1024,  640);
  dense(GOH, 2688, hand_out_w, 1024, hand_out_b, XG + 1024, 2048, Bz, 1024, 2688);

  // 5) final k=3 logits conv over concat(a9, tiled XG) -> (B,T) f32
  logits_kernel<<<Bz * T8, 256, 0, stream>>>(BIG0, XG, logits_w, logits_b, (float*)d_out, T8);
}
